// Encoder_49967649521793
// MI455X (gfx1250) — matrix operand order repaired and checked by reading, not compiled, not run on hardware
//
#include <hip/hip_runtime.h>
#include <hip/hip_bf16.h>

#define HIDDEN 512
#define EMBD   512
#define SEQL   256
#define BATCH  64
#define GATES  2048   // 4*HIDDEN
#define KTOT   (EMBD + HIDDEN)

typedef __attribute__((ext_vector_type(16))) __bf16 v16bf;
typedef __attribute__((ext_vector_type(8)))  float  v8f;
typedef __attribute__((ext_vector_type(4)))  unsigned int u32x4;
typedef __attribute__((ext_vector_type(8)))  int i32x8;
typedef __attribute__((ext_vector_type(4)))  int i32x4;

union BF16x16 { v16bf v; __bf16 e[16]; uint4 q[2]; };
union F32x8   { v8f  v; float  e[8];  };

#if __has_builtin(__builtin_amdgcn_tensor_load_to_lds) && __has_builtin(__builtin_amdgcn_s_wait_tensorcnt)
#define USE_TDM 1
#else
#define USE_TDM 0
#endif

__device__ __forceinline__ unsigned short f2bf(float f) {
  unsigned int u = __float_as_uint(f);
  u += 0x7FFFu + ((u >> 16) & 1u);          // round-to-nearest-even
  return (unsigned short)(u >> 16);
}
__device__ __forceinline__ float sigmoidf_(float x) {
  return 1.0f / (1.0f + __expf(-x));
}

// WMMA A-fragment-major swizzle for a [64 rows][512 k] bf16 matrix.
// Block (kb32*4 + rowblk) holds 32 lanes x 16 contiguous halves.
__device__ __forceinline__ int swzA(int row, int k) {
  int rb = row >> 4, m = row & 15;
  int kb = k >> 5, kl = k & 31;
  int lane = m + (((kl >> 3) & 1) << 4);
  int i = (kl & 7) + ((kl >> 4) << 3);
  return (((kb << 2) + rb) << 9) + (lane << 4) + i;
}
// WMMA B-fragment-major swizzle for the [512 k][2048 n] weight matrix.
// Tile (kb32*128 + ntile16) holds 32 lanes x 16 contiguous halves.
__device__ __forceinline__ int swzB(int k, int n) {
  int kb = k >> 5, kl = k & 31;
  int nt = n >> 4, nl = n & 15;
  int lane = nl + (((kl >> 3) & 1) << 4);
  int i = (kl & 7) + ((kl >> 4) << 3);
  return ((kb * 128 + nt) << 9) + (lane << 4) + i;
}

// ---------------------------------------------------------------- embed gather
// A-swizzled, time-major: Xswz[t*32768 + swzA(b,e)] = bf16(emb[src[b,t], e])
__global__ void embed_kernel(const int* __restrict__ src,
                             const float* __restrict__ emb,
                             unsigned short* __restrict__ Xswz) {
  int idx = blockIdx.x * blockDim.x + threadIdx.x;   // over S*B*E
  int e = idx & (EMBD - 1);
  int b = (idx >> 9) & (BATCH - 1);
  int t = idx >> 15;
  int tok = src[b * SEQL + t];
  Xswz[(t << 15) + swzA(b, e)] = f2bf(emb[tok * EMBD + e]);
}

// ------------------------------------------------- weight convert + B-swizzle
// W: [4H][K] fp32 row-major -> Wt: B-fragment-major bf16
__global__ void wtrans_kernel(const float* __restrict__ W,
                              unsigned short* __restrict__ Wt) {
  int idx = blockIdx.x * blockDim.x + threadIdx.x;   // over 4H*K, coalesced read
  int k = idx & (HIDDEN - 1);
  int n = idx >> 9;
  Wt[swzB(k, n)] = f2bf(W[idx]);
}

// ---------------------------------------------------------------- state init
__global__ void init_kernel(float* c0, float* c1, float* h0f, float* h1f,
                            unsigned short* hb0a, unsigned short* hb0b,
                            unsigned short* hb1a, unsigned short* hb1b) {
  int idx = blockIdx.x * blockDim.x + threadIdx.x;
  if (idx < BATCH * HIDDEN) {
    c0[idx] = 0.f; c1[idx] = 0.f; h0f[idx] = 0.f; h1f[idx] = 0.f;
    hb0a[idx] = 0; hb0b[idx] = 0; hb1a[idx] = 0; hb1b[idx] = 0;
  }
}

// ---------------------------------------------------------------- LSTM step
// One launch = one (layer, timestep). grid = 8 WGs (64 hidden units each),
// block = 256 threads (8 waves). All operands pre-swizzled to fragment layout.
__global__ __launch_bounds__(256)
void lstm_step_kernel(const unsigned short* __restrict__ xswz,   // A-swizzled [64][512]
                      const unsigned short* __restrict__ hswz,   // A-swizzled [64][512]
                      const unsigned short* __restrict__ WIH,    // B-swizzled [512][2048]
                      const unsigned short* __restrict__ WHH,    // B-swizzled [512][2048]
                      const float* __restrict__ bih,
                      const float* __restrict__ bhh,
                      float* __restrict__ cst,                   // [64][512] fp32 in-place
                      float* __restrict__ hfp,                   // [64][512] fp32
                      unsigned short* __restrict__ hnext,        // A-swizzled bf16
                      unsigned short* __restrict__ hseq) {       // A-swizzled bf16 or null
  __shared__ __bf16 ldsW[16 * 512];   // 16 fragment tiles x 1KB = 16KB

  const int tid    = threadIdx.x;
  const int lane   = tid & 31;
  const int w      = tid >> 5;
  const int j0     = blockIdx.x * 64;      // this WG's hidden slice
  const int rowblk = w & 3;                // batch 16-row block
  const int hb     = (w >> 2) * 2;         // two hidden 16-col sub-blocks
  const int nl     = lane & 15;

  v8f acc[2][4];
  #pragma unroll
  for (int hs = 0; hs < 2; ++hs)
    #pragma unroll
    for (int g = 0; g < 4; ++g) acc[hs][g] = (v8f){0.f,0.f,0.f,0.f,0.f,0.f,0.f,0.f};

  for (int kk0 = 0; kk0 < KTOT; kk0 += 32) {
    const unsigned short* Asrc;
    const unsigned short* Wsrc;
    int k;
    if (kk0 < EMBD) { Asrc = xswz; Wsrc = WIH; k = kk0; }
    else            { Asrc = hswz; Wsrc = WHH; k = kk0 - EMBD; }
    const int kb32 = k >> 5;

    __syncthreads();   // previous iteration done reading ldsW

#if USE_TDM
    // TDM: 2D tile = 4 runs (gates) of 4KB at 32KB stride -> 16KB into LDS.
    if (w == 0) {
      unsigned long long ga =
          (unsigned long long)(uintptr_t)(Wsrc + ((size_t)(kb32 * 128 + (j0 >> 4)) << 9));
      u32x4 g0;
      g0[0] = 1u;                                        // count=1, valid, user mode
      g0[1] = (unsigned int)(uintptr_t)&ldsW[0];         // lds_addr (bytes)
      g0[2] = (unsigned int)(ga & 0xFFFFFFFFull);        // global_addr[31:0]
      g0[3] = (unsigned int)((ga >> 32) & 0x1FFFFFFull)  // global_addr[56:32]
              | (2u << 30);                              // type=2 (image)
      i32x8 g1;
      g1[0] = 0x10000;            // workgroup_mask=0, data_size=1 (2 bytes)
      g1[1] = 0;                  // atomic barrier addr=0, tensor_dim0 lo16=0
      g1[2] = 16;                 // tensor_dim0 = 1<<20, tensor_dim1 lo16=0
      g1[3] = 16 | (2048 << 16);  // tensor_dim1 = 1<<20, tile_dim0 = 2048 elems
      g1[4] = 4;                  // tile_dim1 = 4, tile_dim2 = 0
      g1[5] = 16384;              // tensor_dim0_stride = 16384 elems (32KB)
      g1[6] = 0;
      g1[7] = 0;
      i32x4 z4 = (i32x4){0, 0, 0, 0};
      i32x8 z8 = (i32x8){0, 0, 0, 0, 0, 0, 0, 0};
      __builtin_amdgcn_tensor_load_to_lds(g0, g1, z4, z4, z8, 0);
      __builtin_amdgcn_s_wait_tensorcnt(0);
    }
#else
    // Fallback: cooperative fragment-major block copy (pure b128 traffic).
    #pragma unroll
    for (int c = 0; c < 4; ++c) {
      int chunk  = tid * 4 + c;            // 1024 x 16B chunks
      int tl     = chunk >> 6;             // local tile 0..15  (g*4 + hsub)
      int within = (chunk & 63) * 8;       // halves within tile
      int g      = tl >> 2, hsub = tl & 3;
      int gt     = g * 32 + (j0 >> 4) + hsub;
      const uint4 v = *(const uint4*)(Wsrc + ((size_t)(kb32 * 128 + gt) << 9) + within);
      *(uint4*)(&ldsW[(tl << 9) + within]) = v;
    }
#endif
    __syncthreads();

    // A fragment: 32B contiguous per lane, straight from global (L2-hot).
    BF16x16 a;
    {
      const uint4* pa =
          (const uint4*)(Asrc + ((size_t)((kb32 << 2) + rowblk) << 9) + (lane << 4));
      a.q[0] = pa[0];
      a.q[1] = pa[1];
      if (k + 32 < HIDDEN)
        __builtin_prefetch(pa + 128, 0, 1);   // 2 A blocks ahead (+2KB)
    }

    #pragma unroll
    for (int hs = 0; hs < 2; ++hs) {
      #pragma unroll
      for (int g = 0; g < 4; ++g) {
        const int tl = g * 4 + hb + hs;
        BF16x16 bfr;
        const uint4* pb = (const uint4*)(&ldsW[(tl << 9) + (lane << 4)]);
        bfr.q[0] = pb[0];
        bfr.q[1] = pb[1];
        acc[hs][g] = __builtin_amdgcn_wmma_f32_16x16x32_bf16(
            false, a.v, false, bfr.v, (short)0, acc[hs][g], false, false);
      }
    }
  }

  // fused LSTM elementwise: each wave holds i/f/g/o for its (16 batch x 32 hidden)
  const int mofs = (lane >> 4) * 8;
  #pragma unroll
  for (int hs = 0; hs < 2; ++hs) {
    const int j = j0 + (hb + hs) * 16 + nl;
    const float bs0 = bih[0 * HIDDEN + j] + bhh[0 * HIDDEN + j];
    const float bs1 = bih[1 * HIDDEN + j] + bhh[1 * HIDDEN + j];
    const float bs2 = bih[2 * HIDDEN + j] + bhh[2 * HIDDEN + j];
    const float bs3 = bih[3 * HIDDEN + j] + bhh[3 * HIDDEN + j];
    F32x8 ai, af, ag, ao;
    ai.v = acc[hs][0]; af.v = acc[hs][1]; ag.v = acc[hs][2]; ao.v = acc[hs][3];
    #pragma unroll
    for (int r = 0; r < 8; ++r) {
      const int brow = rowblk * 16 + mofs + r;
      const int idx  = brow * HIDDEN + j;
      const float iv = sigmoidf_(ai.e[r] + bs0);
      const float fv = sigmoidf_(af.e[r] + bs1);
      const float gv = tanhf(ag.e[r] + bs2);
      const float ov = sigmoidf_(ao.e[r] + bs3);
      const float ct = fv * cst[idx] + iv * gv;
      const float ht = ov * tanhf(ct);
      cst[idx] = ct;
      hfp[idx] = ht;
      const unsigned short hb16 = f2bf(ht);
      const int soff = swzA(brow, j);
      hnext[soff] = hb16;
      if (hseq) hseq[soff] = hb16;
    }
  }
}

// ---------------------------------------------------------------- output pack
__global__ void finalize_kernel(const float* __restrict__ h0f, const float* __restrict__ h1f,
                                const float* __restrict__ c0,  const float* __restrict__ c1,
                                float* __restrict__ out) {
  int idx = blockIdx.x * blockDim.x + threadIdx.x;   // 4 * B*H
  int seg = idx >> 15;
  int r   = idx & 32767;
  const float* src = (seg == 0) ? h0f : (seg == 1) ? h1f : (seg == 2) ? c0 : c1;
  out[idx] = src[r];
}

extern "C" void kernel_launch(void* const* d_in, const int* in_sizes, int n_in,
                              void* d_out, int out_size, void* d_ws, size_t ws_size,
                              hipStream_t stream) {
  const int*   src   = (const int*)  d_in[0];
  const float* emb   = (const float*)d_in[1];
  const float* W_ih0 = (const float*)d_in[2];
  const float* b_ih0 = (const float*)d_in[3];
  const float* W_hh0 = (const float*)d_in[4];
  const float* b_hh0 = (const float*)d_in[5];
  const float* W_ih1 = (const float*)d_in[6];
  const float* b_ih1 = (const float*)d_in[7];
  const float* W_hh1 = (const float*)d_in[8];
  const float* b_hh1 = (const float*)d_in[9];
  float* out = (float*)d_out;

  char* ws = (char*)d_ws;
  const size_t SZ_X   = (size_t)SEQL * BATCH * EMBD * 2;     // 16 MB
  const size_t SZ_WT  = (size_t)HIDDEN * GATES * 2;          // 2 MB each
  const size_t SZ_HSQ = (size_t)SEQL * BATCH * HIDDEN * 2;   // 16 MB
  const size_t SZ_HB  = (size_t)BATCH * HIDDEN * 2;
  const size_t SZ_F   = (size_t)BATCH * HIDDEN * 4;

  unsigned short* Xbf  = (unsigned short*)(ws);
  unsigned short* Wt0I = (unsigned short*)(ws + SZ_X);
  unsigned short* Wt0H = (unsigned short*)(ws + SZ_X + SZ_WT);
  unsigned short* Wt1I = (unsigned short*)(ws + SZ_X + 2 * SZ_WT);
  unsigned short* Wt1H = (unsigned short*)(ws + SZ_X + 3 * SZ_WT);
  unsigned short* Hseq = (unsigned short*)(ws + SZ_X + 4 * SZ_WT);
  char* p = ws + SZ_X + 4 * SZ_WT + SZ_HSQ;
  unsigned short* hb0a = (unsigned short*)(p);            p += SZ_HB;
  unsigned short* hb0b = (unsigned short*)(p);            p += SZ_HB;
  unsigned short* hb1a = (unsigned short*)(p);            p += SZ_HB;
  unsigned short* hb1b = (unsigned short*)(p);            p += SZ_HB;
  float* c0  = (float*)(p);                               p += SZ_F;
  float* c1  = (float*)(p);                               p += SZ_F;
  float* h0f = (float*)(p);                               p += SZ_F;
  float* h1f = (float*)(p);

  // 1) weights -> bf16, B-fragment-major
  const int wblocks = (GATES * HIDDEN) / 256;
  wtrans_kernel<<<wblocks, 256, 0, stream>>>(W_ih0, Wt0I);
  wtrans_kernel<<<wblocks, 256, 0, stream>>>(W_hh0, Wt0H);
  wtrans_kernel<<<wblocks, 256, 0, stream>>>(W_ih1, Wt1I);
  wtrans_kernel<<<wblocks, 256, 0, stream>>>(W_hh1, Wt1H);
  // 2) embedding gather -> bf16, A-fragment-major, time-major
  embed_kernel<<<(SEQL * BATCH * EMBD) / 256, 256, 0, stream>>>(src, emb, Xbf);
  // 3) zero states
  init_kernel<<<(BATCH * HIDDEN + 255) / 256, 256, 0, stream>>>(
      c0, c1, h0f, h1f, hb0a, hb0b, hb1a, hb1b);

  // 4) recurrence: 2 launches per timestep (layer0 then layer1)
  for (int t = 0; t < SEQL; ++t) {
    unsigned short* h0p = (t & 1) ? hb0b : hb0a;
    unsigned short* h0n = (t & 1) ? hb0a : hb0b;
    unsigned short* h1p = (t & 1) ? hb1b : hb1a;
    unsigned short* h1n = (t & 1) ? hb1a : hb1b;
    const unsigned short* x0 = Xbf  + (size_t)t * BATCH * EMBD;
    unsigned short*       s0 = Hseq + (size_t)t * BATCH * HIDDEN;
    lstm_step_kernel<<<HIDDEN / 64, 256, 0, stream>>>(
        x0, h0p, Wt0I, Wt0H, b_ih0, b_hh0, c0, h0f, h0n, s0);
    lstm_step_kernel<<<HIDDEN / 64, 256, 0, stream>>>(
        s0, h1p, Wt1I, Wt1H, b_ih1, b_hh1, c1, h1f, h1n, (unsigned short*)nullptr);
  }

  // 5) pack [hs(layer0,layer1); cs(layer0,layer1)]
  finalize_kernel<<<(4 * BATCH * HIDDEN) / 256, 256, 0, stream>>>(h0f, h1f, c0, c1, out);
}
